// LSTM_20366734918064
// MI455X (gfx1250) — compile-verified, matrix operand order checked by reading, not run on hardware
//
#include <hip/hip_runtime.h>
#include <hip/hip_bf16.h>
#include <math.h>

// ---------------------------------------------------------------------------
// LSTM on gfx1250 (MI455X): bf16 WMMA recurrent GEMM with fused cell-update
// epilogue (gate-interleaved N layout + quad lane shuffles), fp32 state.
// ---------------------------------------------------------------------------

typedef __attribute__((ext_vector_type(16))) __bf16 v16bf;
typedef __attribute__((ext_vector_type(8)))  float  v8f;

#define SEQ_LEN     512
#define TSTEPS      511          // SEQ_LEN - 1
#define HIDDEN      1024
#define INPUT_DIM   256
#define BATCH       512
#define NUM_CLASSES 10
#define NFUSED      (4 * HIDDEN)            // 4096 fused cols, n = 4*j + gate
#define KFUSED      (INPUT_DIM + HIDDEN)    // 1280 fused K ([x_t ; h])
#define KTILES_X    (INPUT_DIM / 32)        // 8
#define KTILES_H    (HIDDEN / 32)           // 32
#define NTILES      (NFUSED / 16)           // 256

// ---------------------------------------------------------------------------
// Weight packing: bf16, B-fragment-native layout, gate-interleaved columns.
// Fragment (kt, nt) covers K [kt*32, kt*32+32), N [nt*16, nt*16+16).
// Per ISA 7.12.2 (16-bit B 32x16): lane l in [0,16): col=l, K 0..7 & 16..23;
// lane l+16: col=l, K 8..15 & 24..31. Stored as [frag][lane][16 halves].
// Fused column n = 4*j + gate  (j = hidden unit, gate: 0=g,1=i,2=f,3=o).
// ---------------------------------------------------------------------------
__global__ void pack_weights_kernel(const float* __restrict__ Wgx, const float* __restrict__ Wgh,
                                    const float* __restrict__ Wix, const float* __restrict__ Wih,
                                    const float* __restrict__ Wfx, const float* __restrict__ Wfh,
                                    const float* __restrict__ Wox, const float* __restrict__ Woh,
                                    __bf16* __restrict__ Wpack) {
    int e = blockIdx.x * blockDim.x + threadIdx.x;
    if (e >= KFUSED * NFUSED) return;
    int h    = e & 15;          // half index within lane
    int lane = (e >> 4) & 31;
    int frag = e >> 9;          // kt*NTILES + nt
    int nt   = frag & (NTILES - 1);
    int kt   = frag >> 8;
    int nin  = lane & 15;
    int kin  = (h & 7) | ((lane >> 4) << 3) | ((h >> 3) << 4);
    int k = kt * 32 + kin;
    int n = nt * 16 + nin;
    int gate = n & 3;           // interleaved: 0..3 -> g,i,f,o
    int col  = n >> 2;          // hidden unit j
    const float* Wx = (gate == 0) ? Wgx : (gate == 1) ? Wix : (gate == 2) ? Wfx : Wox;
    const float* Wh = (gate == 0) ? Wgh : (gate == 1) ? Wih : (gate == 2) ? Wfh : Woh;
    float v = (k < INPUT_DIM) ? Wx[k * HIDDEN + col]
                              : Wh[(k - INPUT_DIM) * HIDDEN + col];
    Wpack[e] = (__bf16)v;
}

// Fused interleaved bias: bfused[4*j + gate] = b_gate[j]
__global__ void pack_bias_kernel(const float* __restrict__ b_g, const float* __restrict__ b_i,
                                 const float* __restrict__ b_f, const float* __restrict__ b_o,
                                 float* __restrict__ bfused) {
    int n = blockIdx.x * blockDim.x + threadIdx.x;
    if (n >= NFUSED) return;
    int gate = n & 3, j = n >> 2;
    const float* b = (gate == 0) ? b_g : (gate == 1) ? b_i : (gate == 2) ? b_f : b_o;
    bfused[n] = b[j];
}

// Convert embedding table fp32 -> bf16 (gather happens inside the GEMM).
__global__ void convert_emb_kernel(const float* __restrict__ emb, __bf16* __restrict__ emb_bf) {
    int e = blockIdx.x * blockDim.x + threadIdx.x;
    if (e < SEQ_LEN * INPUT_DIM) emb_bf[e] = (__bf16)emb[e];
}

// h0 = 0 (bf16), c0 = 0 (fp32)
__global__ void init_state_kernel(__bf16* __restrict__ h_bf, float* __restrict__ c) {
    int e = blockIdx.x * blockDim.x + threadIdx.x;
    if (e < BATCH * HIDDEN) { h_bf[e] = (__bf16)0.0f; c[e] = 0.0f; }
}

// ---------------------------------------------------------------------------
// Fused per-step kernel:  pre = [emb[x[:,t]] | h_in] @ Wpack + bfused
// followed in-register by the LSTM cell update:
//   c = tanh(pg)*sig(pi) + c*sig(pf);  h_out = tanh(c)*sig(po)
// Block = 256 threads = 8 waves, block tile 64(M) x 128(N),
// waves 2(M) x 4(N), each wave a 32x32 tile = 2x2 wmma_f32_16x16x32_bf16.
// A wave's 32 columns = 8 hidden units x 4 gates (gate = lane&3).
// ---------------------------------------------------------------------------
union AFrag { v16bf v; uint4 q[2]; };

__global__ __launch_bounds__(256) void lstm_step_kernel(
        const int*    __restrict__ x,       // [BATCH][SEQ_LEN]
        const __bf16* __restrict__ emb_bf,  // [SEQ_LEN][INPUT_DIM]
        const __bf16* __restrict__ h_in,    // [BATCH][HIDDEN]
        const __bf16* __restrict__ Wpack,   // packed fragments
        const float*  __restrict__ bfused,  // [NFUSED]
        float*        __restrict__ c,       // [BATCH][HIDDEN] fp32 cell state
        __bf16*       __restrict__ h_out,   // [BATCH][HIDDEN]
        int t) {
    const int lane  = threadIdx.x & 31;
    const int wid   = threadIdx.x >> 5;
    const int wm    = wid >> 2;             // 0..1
    const int wn    = wid & 3;              // 0..3
    const int Mbase = blockIdx.y * 64 + wm * 32;
    const int Nbase = blockIdx.x * 128 + wn * 32;
    const int rlane = lane & 15;            // A row / B-C col within 16
    const int khalf = lane >> 4;            // 0: K 0..7 & 16..23, 1: K 8..15 & 24..31

    // Per-lane A row pointers for the two M-fragments (embedding gather here).
    const __bf16* arow_x[2];
    const __bf16* arow_h[2];
#pragma unroll
    for (int fm = 0; fm < 2; ++fm) {
        int row = Mbase + fm * 16 + rlane;          // batch index
        int tok = x[row * SEQ_LEN + t];
        arow_x[fm] = emb_bf + (size_t)tok * INPUT_DIM;
        arow_h[fm] = h_in  + (size_t)row * HIDDEN;
    }

    v8f acc[2][2] = {};

    // ---- K region 1: x-projection (K tiles 0..7), no pointer selects ----
    for (int kt = 0; kt < KTILES_X; ++kt) {
        const int k0 = kt * 32;
        AFrag a[2];
#pragma unroll
        for (int fm = 0; fm < 2; ++fm) {
            const __bf16* rp = arow_x[fm] + k0;
            a[fm].q[0] = *(const uint4*)(rp + khalf * 8);
            a[fm].q[1] = *(const uint4*)(rp + 16 + khalf * 8);
        }
        AFrag b[2];
#pragma unroll
        for (int fn = 0; fn < 2; ++fn) {
            int ntile = (Nbase >> 4) + fn;
            const __bf16* bp = Wpack + (((size_t)kt * NTILES + ntile) * 32 + lane) * 16;
            b[fn].q[0] = *(const uint4*)(bp);
            b[fn].q[1] = *(const uint4*)(bp + 8);
        }
#pragma unroll
        for (int fm = 0; fm < 2; ++fm)
#pragma unroll
            for (int fn = 0; fn < 2; ++fn)
                acc[fm][fn] = __builtin_amdgcn_wmma_f32_16x16x32_bf16(
                    false, a[fm].v, false, b[fn].v,
                    (short)0, acc[fm][fn], false, false);
    }

    // ---- K region 2: h-projection (K tiles 8..39) ----
    for (int kt = 0; kt < KTILES_H; ++kt) {
        const int k0 = kt * 32;
        AFrag a[2];
#pragma unroll
        for (int fm = 0; fm < 2; ++fm) {
            const __bf16* rp = arow_h[fm] + k0;
            a[fm].q[0] = *(const uint4*)(rp + khalf * 8);
            a[fm].q[1] = *(const uint4*)(rp + 16 + khalf * 8);
        }
        AFrag b[2];
#pragma unroll
        for (int fn = 0; fn < 2; ++fn) {
            int ntile = (Nbase >> 4) + fn;
            const __bf16* bp = Wpack +
                ((((size_t)kt + KTILES_X) * NTILES + ntile) * 32 + lane) * 16;
            b[fn].q[0] = *(const uint4*)(bp);
            b[fn].q[1] = *(const uint4*)(bp + 8);
        }
#pragma unroll
        for (int fm = 0; fm < 2; ++fm)
#pragma unroll
            for (int fn = 0; fn < 2; ++fn)
                acc[fm][fn] = __builtin_amdgcn_wmma_f32_16x16x32_bf16(
                    false, a[fm].v, false, b[fn].v,
                    (short)0, acc[fm][fn], false, false);
    }

    // ---- Fused epilogue: bias + quad shuffle + LSTM cell update ----
    // C/D layout: lane l<16 -> col=l, rows 0..7 (VGPR r = row r); l>=16 -> rows 8..15.
    // Within each lane quad: gate = lane&3; the 4 gates of hidden unit j sit in
    // lanes quad^{0,1,2,3}. shfl_xor(1/2/3) hands the gate-0 lane pi/pf/po.
    float biasv[2];
#pragma unroll
    for (int fn = 0; fn < 2; ++fn)
        biasv[fn] = bfused[Nbase + fn * 16 + rlane];

    const int gate = rlane & 3;
#pragma unroll
    for (int fm = 0; fm < 2; ++fm) {
#pragma unroll
        for (int fn = 0; fn < 2; ++fn) {
            const int col   = Nbase + fn * 16 + rlane;
            const int j     = col >> 2;                     // hidden unit
            const int rbase = Mbase + fm * 16 + 8 * khalf;  // first row of this half
#pragma unroll
            for (int r = 0; r < 8; ++r) {
                float v0 = acc[fm][fn][r] + biasv[fn];
                float v1 = __shfl_xor(v0, 1, 32);
                float v2 = __shfl_xor(v0, 2, 32);
                float v3 = __shfl_xor(v0, 3, 32);
                if (gate == 0) {
                    // gate-0 lane: v0=pg, v1=pi, v2=pf, v3=po
                    const int row = rbase + r;
                    const size_t idx = (size_t)row * HIDDEN + j;
                    float gg = tanhf(v0);
                    float ig = 1.0f / (1.0f + __expf(-v1));
                    float fg = 1.0f / (1.0f + __expf(-v2));
                    float og = 1.0f / (1.0f + __expf(-v3));
                    float cn = gg * ig + c[idx] * fg;
                    c[idx] = cn;
                    h_out[idx] = (__bf16)(tanhf(cn) * og);
                }
            }
        }
    }
}

// ---------------------------------------------------------------------------
// Head: logits = h @ W_ph + b_p ; out = log_softmax(logits). Tiny: 1 thr / row.
// ---------------------------------------------------------------------------
__global__ void head_kernel(const __bf16* __restrict__ h_bf,
                            const float* __restrict__ W_ph,
                            const float* __restrict__ b_p,
                            float* __restrict__ out) {
    int b = blockIdx.x * blockDim.x + threadIdx.x;
    if (b >= BATCH) return;
    float logit[NUM_CLASSES];
#pragma unroll
    for (int j = 0; j < NUM_CLASSES; ++j) logit[j] = b_p[j];
    const __bf16* hr = h_bf + (size_t)b * HIDDEN;
    for (int k = 0; k < HIDDEN; ++k) {
        float hv = (float)hr[k];
        const float* wr = W_ph + (size_t)k * NUM_CLASSES;
#pragma unroll
        for (int j = 0; j < NUM_CLASSES; ++j) logit[j] += hv * wr[j];
    }
    float mx = logit[0];
#pragma unroll
    for (int j = 1; j < NUM_CLASSES; ++j) mx = fmaxf(mx, logit[j]);
    float s = 0.0f;
#pragma unroll
    for (int j = 0; j < NUM_CLASSES; ++j) s += __expf(logit[j] - mx);
    float lse = mx + __logf(s);
#pragma unroll
    for (int j = 0; j < NUM_CLASSES; ++j) out[b * NUM_CLASSES + j] = logit[j] - lse;
}

// ---------------------------------------------------------------------------
// Launch
// ---------------------------------------------------------------------------
extern "C" void kernel_launch(void* const* d_in, const int* in_sizes, int n_in,
                              void* d_out, int out_size, void* d_ws, size_t ws_size,
                              hipStream_t stream) {
    const int*   x    = (const int*)  d_in[0];
    const float* emb  = (const float*)d_in[1];
    const float* Wgx  = (const float*)d_in[2];
    const float* Wgh  = (const float*)d_in[3];
    const float* b_g  = (const float*)d_in[4];
    const float* Wix  = (const float*)d_in[5];
    const float* Wih  = (const float*)d_in[6];
    const float* b_i  = (const float*)d_in[7];
    const float* Wfx  = (const float*)d_in[8];
    const float* Wfh  = (const float*)d_in[9];
    const float* b_f  = (const float*)d_in[10];
    const float* Wox  = (const float*)d_in[11];
    const float* Woh  = (const float*)d_in[12];
    const float* b_o  = (const float*)d_in[13];
    const float* W_ph = (const float*)d_in[14];
    const float* b_p  = (const float*)d_in[15];
    float* out = (float*)d_out;

    // Workspace layout (256B aligned), ~14.8 MB total
    char* ws = (char*)d_ws;
    size_t off = 0;
    auto alloc = [&](size_t bytes) {
        char* p = ws + off;
        off += (bytes + 255) & ~(size_t)255;
        return p;
    };
    __bf16* Wpack  = (__bf16*)alloc((size_t)KFUSED * NFUSED * sizeof(__bf16)); // 10.5 MB
    __bf16* emb_bf = (__bf16*)alloc((size_t)SEQ_LEN * INPUT_DIM * sizeof(__bf16));
    float*  bfused = (float*) alloc((size_t)NFUSED * sizeof(float));
    __bf16* hA     = (__bf16*)alloc((size_t)BATCH * HIDDEN * sizeof(__bf16));
    __bf16* hB     = (__bf16*)alloc((size_t)BATCH * HIDDEN * sizeof(__bf16));
    float*  c      = (float*) alloc((size_t)BATCH * HIDDEN * sizeof(float));
    (void)ws_size;

    // One-time prep (re-run every call: deterministic, inputs untouched)
    {
        int n = KFUSED * NFUSED;
        pack_weights_kernel<<<(n + 255) / 256, 256, 0, stream>>>(
            Wgx, Wgh, Wix, Wih, Wfx, Wfh, Wox, Woh, Wpack);
    }
    pack_bias_kernel<<<(NFUSED + 255) / 256, 256, 0, stream>>>(b_g, b_i, b_f, b_o, bfused);
    {
        int n = SEQ_LEN * INPUT_DIM;
        convert_emb_kernel<<<(n + 255) / 256, 256, 0, stream>>>(emb, emb_bf);
    }
    {
        int n = BATCH * HIDDEN;
        init_state_kernel<<<(n + 255) / 256, 256, 0, stream>>>(hA, c);
    }

    // Recurrence: fused GEMM + cell update, h double-buffered.
    dim3 ggrid(NFUSED / 128, BATCH / 64);   // (32, 8)
    __bf16* h_in = hA;
    __bf16* h_out = hB;
    for (int t = 0; t < TSTEPS; ++t) {
        lstm_step_kernel<<<ggrid, 256, 0, stream>>>(
            x, emb_bf, h_in, Wpack, bfused, c, h_out, t);
        __bf16* tmp = h_in; h_in = h_out; h_out = tmp;
    }

    // Head + log_softmax on final h (h_in after the last swap)
    head_kernel<<<(BATCH + 255) / 256, 256, 0, stream>>>(h_in, W_ph, b_p, out);
    (void)in_sizes; (void)n_in; (void)out_size;
}
